// NanoGPT_69226282877409
// MI455X (gfx1250) — matrix operand order repaired and checked by reading, not compiled, not run on hardware
//
#include <hip/hip_runtime.h>

// ---------------- constants (shapes fixed by the reference) ----------------
#define NELEM   1048576      // B*T*E = 2*512*1024
#define SCALE_A 0.03125f     // 1/sqrt(E) = 1/32

typedef __attribute__((ext_vector_type(16))) __bf16 v16bf;
typedef __attribute__((ext_vector_type(8)))  float  v8f;

union FragAB { v16bf v; uint4 q[2]; };
union U4     { uint4 q; unsigned short u[8]; };

__device__ __forceinline__ unsigned int f2bf1(float f) {
  unsigned int u = __float_as_uint(f);
  return (u + 0x7fffu + ((u >> 16) & 1u)) >> 16;   // RNE fp32 -> bf16
}
__device__ __forceinline__ unsigned int pack2bf(float a, float b) {
  return f2bf1(a) | (f2bf1(b) << 16);
}
__device__ __forceinline__ unsigned short f2bf(float f) {
  return (unsigned short)f2bf1(f);
}
__device__ __forceinline__ v8f wmma_bf16(FragAB a, FragAB b, v8f c) {
  return __builtin_amdgcn_wmma_f32_16x16x32_bf16(false, a.v, false, b.v,
                                                 (short)0, c, false, false);
}

// async 16B copy global -> LDS (CDNA5 GLOBAL_LOAD_ASYNC_TO_LDS_B128, ASYNCcnt)
// LDS aperture: flat addr[31:0] == LDS byte address, so truncation is the offset.
__device__ __forceinline__ void async_b128(const void* lds_ptr, const void* gptr) {
  unsigned off = (unsigned)(size_t)lds_ptr;
  asm volatile("global_load_async_to_lds_b128 %0, %1, off"
               :: "v"(off), "v"(gptr) : "memory");
}
__device__ __forceinline__ void wait_async0() {
  asm volatile("s_wait_asynccnt 0" ::: "memory");
}

// ---------------- embedding + sinusoidal PE (fp32 residual stream) ----------
__global__ void embed_k(const int* __restrict__ tok, const float* __restrict__ emb,
                        float* __restrict__ x) {
  int idx = blockIdx.x * 256 + threadIdx.x;          // < 2^20
  int e = idx & 1023;
  int t = (idx >> 10) & 511;
  int b = idx >> 19;
  int tk = tok[b * 512 + t];
  float base = emb[(size_t)tk * 1024 + e];
  float div = __expf((float)(e & ~1) * (-9.210340371976184f / 1024.f));
  float ang = (float)t * div;
  x[idx] = base + ((e & 1) ? __cosf(ang) : __sinf(ang));
}

// ---------------- global RMS (mean over whole tensor) ----------------
__global__ void rms_part_k(const float* __restrict__ x, float* __restrict__ part) {
  __shared__ float sm[256];
  float s = 0.f;
  for (int i = blockIdx.x * 256 + threadIdx.x; i < NELEM; i += 1024 * 256) {
    float v = x[i]; s += v * v;
  }
  sm[threadIdx.x] = s; __syncthreads();
  for (int st = 128; st > 0; st >>= 1) {
    if (threadIdx.x < st) sm[threadIdx.x] += sm[threadIdx.x + st];
    __syncthreads();
  }
  if (threadIdx.x == 0) part[blockIdx.x] = sm[0];
}

__global__ void rms_fin_k(const float* __restrict__ part, float* __restrict__ scal) {
  __shared__ float sm[256];
  float s = 0.f;
  for (int i = threadIdx.x; i < 1024; i += 256) s += part[i];
  sm[threadIdx.x] = s; __syncthreads();
  for (int st = 128; st > 0; st >>= 1) {
    if (threadIdx.x < st) sm[threadIdx.x] += sm[threadIdx.x + st];
    __syncthreads();
  }
  if (threadIdx.x == 0) scal[0] = sqrtf(sm[0] / (float)NELEM + 1e-5f);
}

// normalize fp32 residual -> bf16 activations (packed dword stores)
__global__ void rms_norm_k(const float* __restrict__ x, const float* __restrict__ g,
                           const float* __restrict__ scal, unsigned int* __restrict__ y) {
  int idx = blockIdx.x * 256 + threadIdx.x;   // NELEM/2 pairs
  float inv = 1.f / scal[0];
  int i0 = idx * 2;
  float a = x[i0]     * inv * g[i0 & 1023];
  float b = x[i0 + 1] * inv * g[(i0 + 1) & 1023];
  y[idx] = pack2bf(a, b);
}

// ---------------- tiled bf16 WMMA GEMM ----------------
// C[1024, N] = A_bf16[1024, K](lda) @ B_f32[K, N]  (+bias, +epilogue)
// EPI 0: bias   EPI 1: bias + residual(aux, stride ldc)   EPI 2: silu(aux)*(acc+bias)
// OUTBF: 1 -> store bf16, 0 -> store fp32
template<int EPI, int OUTBF>
__global__ __launch_bounds__(256) void gemm_k(
    const unsigned short* __restrict__ A, const float* __restrict__ Bw,
    const float* __restrict__ bias, const float* aux, void* Cv,
    int N, int K, int lda, int ldc) {
  __shared__ unsigned int la[128][20];   // 128 x 32 bf16, padded rows (80B)
  __shared__ unsigned int lb[128][20];   // [n][k] transposed, 128 x 32 bf16

  const int tid = threadIdx.x;
  const int lane = tid & 31, w = tid >> 5;
  const int half = lane >> 4, l15 = lane & 15;
  const int wm = w >> 2, wn = w & 3;             // 2 x 4 wave grid
  const int m0 = blockIdx.y << 7, n0 = blockIdx.x << 7;

  const v8f z8 = {0.f,0.f,0.f,0.f,0.f,0.f,0.f,0.f};
  v8f acc[4][2];
  #pragma unroll
  for (int i = 0; i < 4; ++i)
    #pragma unroll
    for (int j = 0; j < 2; ++j) acc[i][j] = z8;

  const int nk = (K + 31) >> 5;
  for (int kt = 0; kt < nk; ++kt) {
    const int k0 = kt << 5;
    // A tile: async copy 128x32 bf16 (8KB) global -> LDS, overlapped with B work
    #pragma unroll
    for (int i = 0; i < 2; ++i) {
      int c = tid + (i << 8);
      int row = c >> 2, seg = c & 3;
      const unsigned short* g = A + (size_t)(m0 + row) * lda + k0 + seg * 8;
      async_b128(&la[row][seg * 4], g);
    }
    // B tile: 32x128 f32 -> bf16, transposed [n][k] in LDS
    unsigned short* lbs = (unsigned short*)&lb[0][0];
    #pragma unroll
    for (int i = 0; i < 4; ++i) {
      int f = tid + (i << 8);
      int kr = f >> 5, c4 = f & 31;
      int gk = k0 + kr, gn = n0 + (c4 << 2);
      const float* s = Bw + (size_t)gk * N + gn;
      bool kv = gk < K;
      float x0 = (kv && gn     < N) ? s[0] : 0.f;
      float x1 = (kv && gn + 1 < N) ? s[1] : 0.f;
      float x2 = (kv && gn + 2 < N) ? s[2] : 0.f;
      float x3 = (kv && gn + 3 < N) ? s[3] : 0.f;
      if (i == 0 && gk + 32 < K) __builtin_prefetch(s + (size_t)32 * N, 0, 1);
      lbs[(c4 * 4 + 0) * 40 + kr] = f2bf(x0);
      lbs[(c4 * 4 + 1) * 40 + kr] = f2bf(x1);
      lbs[(c4 * 4 + 2) * 40 + kr] = f2bf(x2);
      lbs[(c4 * 4 + 3) * 40 + kr] = f2bf(x3);
    }
    wait_async0();
    __syncthreads();

    FragAB af[4], bf[2];
    #pragma unroll
    for (int mt = 0; mt < 4; ++mt) {               // A: K = half*16 + e
      int r = wm * 64 + mt * 16 + l15;
      af[mt].q[0] = *(const uint4*)&la[r][half * 8];
      af[mt].q[1] = *(const uint4*)&la[r][half * 8 + 4];
    }
    #pragma unroll
    for (int nt = 0; nt < 2; ++nt) {               // B: lane=N, K = half*16 + e
      int r = wn * 32 + nt * 16 + l15;
      bf[nt].q[0] = *(const uint4*)&lb[r][half * 8];
      bf[nt].q[1] = *(const uint4*)&lb[r][half * 8 + 4];
    }
    #pragma unroll
    for (int mt = 0; mt < 4; ++mt)
      #pragma unroll
      for (int nt = 0; nt < 2; ++nt)
        acc[mt][nt] = wmma_bf16(af[mt], bf[nt], acc[mt][nt]);
    __syncthreads();
  }

  // epilogue: C layout -> VGPR r: rows r (lanes 0-15) / r+8 (lanes 16-31)
  float* Cf = (float*)Cv;
  unsigned short* Cb = (unsigned short*)Cv;
  #pragma unroll
  for (int mt = 0; mt < 4; ++mt) {
    int rowb = m0 + wm * 64 + mt * 16 + half * 8;
    #pragma unroll
    for (int nt = 0; nt < 2; ++nt) {
      int col = n0 + wn * 32 + nt * 16 + l15;
      if (col < N) {
        float bv = bias[col];
        #pragma unroll
        for (int r = 0; r < 8; ++r) {
          size_t idx = (size_t)(rowb + r) * ldc + col;
          float v = acc[mt][nt][r] + bv;
          if (EPI == 1) v += aux[idx];
          if (EPI == 2) { float g = aux[idx]; v *= g / (1.f + __expf(-g)); }
          if (OUTBF) Cb[idx] = f2bf(v); else Cf[idx] = v;
        }
      }
    }
  }
}

// ---------------- flash-style causal attention (bf16 qkv in, bf16 out) -------
// grid (T/128, H, B); 8 waves x 16 q-rows; K,V^T chunks of 128 keys in LDS
__global__ __launch_bounds__(256) void attn_k(const unsigned short* __restrict__ qkvb,
                                              unsigned short* __restrict__ out) {
  __shared__ unsigned int kl[128][36];     // K chunk [key][s] bf16 (async-filled)
  __shared__ unsigned int vt[64][68];      // V^T chunk [s][key] bf16
  __shared__ unsigned int pl[8][16][20];   // per-wave P 16x32 bf16

  const int tid = threadIdx.x;
  const int lane = tid & 31, w = tid >> 5;
  const int half = lane >> 4, l15 = lane & 15;
  const int qb = blockIdx.x << 7;
  const int h = blockIdx.y, b = blockIdx.z;
  const int qlo = qb + w * 16;
  const v8f z8 = {0.f,0.f,0.f,0.f,0.f,0.f,0.f,0.f};

  // Q fragments: direct 16B bf16 loads, already in A-fragment layout
  const unsigned short* qr = qkvb + (size_t)(b * 512 + qlo + l15) * 3072 + h * 192;
  FragAB qf[2];
  #pragma unroll
  for (int f = 0; f < 2; ++f) {
    qf[f].q[0] = *(const uint4*)(qr + f * 32 + half * 16);
    qf[f].q[1] = *(const uint4*)(qr + f * 32 + half * 16 + 8);
  }

  v8f o[4];
  #pragma unroll
  for (int i = 0; i < 4; ++i) o[i] = z8;
  float mrow[8], lrow[8];
  #pragma unroll
  for (int r = 0; r < 8; ++r) { mrow[r] = -1e30f; lrow[r] = 0.f; }

  const int nch = blockIdx.x + 1;                    // causal chunk bound
  for (int c = 0; c < nch; ++c) {
    const int kbase = c << 7;
    unsigned short* vts = (unsigned short*)&vt[0][0];
    #pragma unroll
    for (int i = 0; i < 4; ++i) {                    // K: async copy; V: transpose
      int cc = tid + (i << 8);
      int row = cc >> 3, seg = cc & 7;
      const unsigned short* gk_ = qkvb + (size_t)(b * 512 + kbase + row) * 3072
                                       + h * 192 + 64 + seg * 8;
      async_b128(&kl[row][seg * 4], gk_);
      U4 xv; xv.q = *(const uint4*)(gk_ + 64);       // V row chunk (8 bf16)
      #pragma unroll
      for (int j = 0; j < 8; ++j)
        vts[(seg * 8 + j) * 136 + row] = xv.u[j];
    }
    wait_async0();
    __syncthreads();

    #pragma unroll
    for (int ss = 0; ss < 4; ++ss) {                 // 32 keys per sub-step
      int kb = kbase + ss * 32;
      if (kb <= qlo + 15) {                          // wave-uniform causal skip
        v8f sa[2];
        #pragma unroll
        for (int nt = 0; nt < 2; ++nt) {             // S = Q @ K^T
          int kr = ss * 32 + nt * 16 + l15;
          FragAB kb0, kb1;
          kb0.q[0] = *(const uint4*)&kl[kr][half * 8];
          kb0.q[1] = *(const uint4*)&kl[kr][half * 8 + 4];
          kb1.q[0] = *(const uint4*)&kl[kr][16 + half * 8];
          kb1.q[1] = *(const uint4*)&kl[kr][16 + half * 8 + 4];
          v8f t = z8;
          t = wmma_bf16(qf[0], kb0, t);
          t = wmma_bf16(qf[1], kb1, t);
          sa[nt] = t;
        }
        float alpha[8], p0[8], p1[8];
        #pragma unroll
        for (int r = 0; r < 8; ++r) {                // online softmax
          int q = qlo + r + half * 8;
          int k0i = kb + l15, k1i = kb + 16 + l15;
          float s0v = (k0i <= q) ? sa[0][r] * SCALE_A : -1e30f;
          float s1v = (k1i <= q) ? sa[1][r] * SCALE_A : -1e30f;
          float mx = fmaxf(s0v, s1v);
          mx = fmaxf(mx, __shfl_xor(mx, 1, 32));
          mx = fmaxf(mx, __shfl_xor(mx, 2, 32));
          mx = fmaxf(mx, __shfl_xor(mx, 4, 32));
          mx = fmaxf(mx, __shfl_xor(mx, 8, 32));
          float nm = fmaxf(mrow[r], mx);
          float a  = __expf(mrow[r] - nm);
          float e0 = (k0i <= q) ? __expf(s0v - nm) : 0.f;
          float e1 = (k1i <= q) ? __expf(s1v - nm) : 0.f;
          float ps = e0 + e1;
          ps += __shfl_xor(ps, 1, 32);
          ps += __shfl_xor(ps, 2, 32);
          ps += __shfl_xor(ps, 4, 32);
          ps += __shfl_xor(ps, 8, 32);
          lrow[r] = lrow[r] * a + ps;
          mrow[r] = nm;
          alpha[r] = a;
          p0[r] = e0; p1[r] = e1;
        }
        // C-layout -> A-layout transpose of P via per-wave LDS
        unsigned short* ps16 = (unsigned short*)&pl[w][0][0];
        #pragma unroll
        for (int r = 0; r < 8; ++r) {
          int pr = r + half * 8;
          ps16[pr * 40 + l15]      = f2bf(p0[r]);
          ps16[pr * 40 + 16 + l15] = f2bf(p1[r]);
        }
        asm volatile("s_wait_dscnt 0" ::: "memory"); // intra-wave cross-lane dep
        FragAB pf;
        pf.q[0] = *(const uint4*)&pl[w][l15][half * 8];
        pf.q[1] = *(const uint4*)&pl[w][l15][half * 8 + 4];
        #pragma unroll
        for (int nt = 0; nt < 4; ++nt) {             // O = O*alpha + P @ V
          FragAB vf;
          int sr = nt * 16 + l15;
          vf.q[0] = *(const uint4*)&vt[sr][ss * 16 + half * 8];
          vf.q[1] = *(const uint4*)&vt[sr][ss * 16 + half * 8 + 4];
          v8f oo = o[nt];
          #pragma unroll
          for (int r = 0; r < 8; ++r) oo[r] *= alpha[r];
          o[nt] = wmma_bf16(pf, vf, oo);
        }
      }
    }
    __syncthreads();
  }

  float inv[8];
  #pragma unroll
  for (int r = 0; r < 8; ++r) inv[r] = 1.f / lrow[r];
  #pragma unroll
  for (int nt = 0; nt < 4; ++nt) {
    int col = h * 64 + nt * 16 + l15;
    #pragma unroll
    for (int r = 0; r < 8; ++r) {
      int t = qlo + r + half * 8;
      out[(size_t)(b * 512 + t) * 1024 + col] = f2bf(o[nt][r] * inv[r]);
    }
  }
}

// ---------------- driver ----------------
extern "C" void kernel_launch(void* const* d_in, const int* in_sizes, int n_in,
                              void* d_out, int out_size, void* d_ws, size_t ws_size,
                              hipStream_t stream) {
  (void)in_sizes; (void)n_in; (void)out_size; (void)ws_size;
  const int*   tokens = (const int*)d_in[0];
  const float* emb    = (const float*)d_in[1];
  const float* Wqkv   = (const float*)d_in[2];
  const float* bqkv   = (const float*)d_in[3];
  const float* Wo     = (const float*)d_in[4];
  const float* bo     = (const float*)d_in[5];
  const float* W1     = (const float*)d_in[6];
  const float* b1     = (const float*)d_in[7];
  const float* W3     = (const float*)d_in[8];
  const float* b3     = (const float*)d_in[9];
  const float* W2     = (const float*)d_in[10];
  const float* b2     = (const float*)d_in[11];
  const float* gmha   = (const float*)d_in[12];
  const float* gff    = (const float*)d_in[13];
  const float* gfin   = (const float*)d_in[14];
  const float* Wout   = (const float*)d_in[15];
  const float* bout   = (const float*)d_in[16];

  // workspace carve (byte offsets, all 16B-aligned)
  char* base = (char*)d_ws;
  float*          x     = (float*)(base);                      // fp32 residual [1024,1024]
  unsigned int*   hbufw = (unsigned int*)(base + 4194304);     // bf16 normed acts (dword view)
  unsigned short* hbuf  = (unsigned short*)(base + 4194304);
  unsigned short* qkvb  = (unsigned short*)(base + 6291456);   // bf16 [1024,3072]
  unsigned short* attnb = (unsigned short*)(base + 12582912);  // bf16 [1024,1024]
  float*          u1    = (float*)(base + 14680064);           // fp32 [1024,2736]
  unsigned short* u2b   = (unsigned short*)(base + 25886720);  // bf16 [1024,2736]
  float*          part  = (float*)(base + 31490048);
  float*          scal  = (float*)(base + 31494144);

  embed_k<<<NELEM / 256, 256, 0, stream>>>(tokens, emb, x);

  for (int l = 0; l < 24; ++l) {
    const float* wqkv_l = Wqkv + (size_t)l * 1024 * 3072;
    const float* bqkv_l = bqkv + (size_t)l * 3072;
    const float* wo_l   = Wo   + (size_t)l * 1024 * 1024;
    const float* bo_l   = bo   + (size_t)l * 1024;
    const float* w1_l   = W1   + (size_t)l * 1024 * 2730;
    const float* b1_l   = b1   + (size_t)l * 2730;
    const float* w3_l   = W3   + (size_t)l * 1024 * 2730;
    const float* b3_l   = b3   + (size_t)l * 2730;
    const float* w2_l   = W2   + (size_t)l * 2730 * 1024;
    const float* b2_l   = b2   + (size_t)l * 1024;

    // --- MHA ---
    rms_part_k<<<1024, 256, 0, stream>>>(x, part);
    rms_fin_k<<<1, 256, 0, stream>>>(part, scal);
    rms_norm_k<<<NELEM / 512, 256, 0, stream>>>(x, gmha + l * 1024, scal, hbufw);

    gemm_k<0,1><<<dim3(24, 8), 256, 0, stream>>>(hbuf, wqkv_l, bqkv_l, nullptr,
                                                 qkvb, 3072, 1024, 1024, 3072);
    attn_k<<<dim3(4, 16, 2), 256, 0, stream>>>(qkvb, attnb);
    gemm_k<1,0><<<dim3(8, 8), 256, 0, stream>>>(attnb, wo_l, bo_l, x,
                                                x, 1024, 1024, 1024, 1024);

    // --- SwiGLU FFN ---
    rms_part_k<<<1024, 256, 0, stream>>>(x, part);
    rms_fin_k<<<1, 256, 0, stream>>>(part, scal);
    rms_norm_k<<<NELEM / 512, 256, 0, stream>>>(x, gff + l * 1024, scal, hbufw);

    gemm_k<0,0><<<dim3(22, 8), 256, 0, stream>>>(hbuf, w1_l, b1_l, nullptr,
                                                 u1, 2730, 1024, 1024, 2736);
    gemm_k<2,1><<<dim3(22, 8), 256, 0, stream>>>(hbuf, w3_l, b3_l, u1,
                                                 u2b, 2730, 1024, 1024, 2736);
    gemm_k<1,0><<<dim3(8, 8), 256, 0, stream>>>(u2b, w2_l, b2_l, x,
                                                x, 1024, 2730, 2736, 1024);
  }

  // --- final norm + logits ---
  rms_part_k<<<1024, 256, 0, stream>>>(x, part);
  rms_fin_k<<<1, 256, 0, stream>>>(part, scal);
  rms_norm_k<<<NELEM / 512, 256, 0, stream>>>(x, gfin, scal, hbufw);

  gemm_k<0,0><<<dim3(64, 8), 256, 0, stream>>>(hbuf, Wout, bout, nullptr,
                                               (float*)d_out, 8124, 1024, 1024, 8124);
}
